// UnifiedMemoryLayer_25486335934751
// MI455X (gfx1250) — matrix operand order repaired and checked by reading, not compiled, not run on hardware
//
#include <hip/hip_runtime.h>
#include <hip/hip_bf16.h>

// ---------------- problem constants ----------------
#define B_   2
#define S_   2048
#define D_   1024
#define H_   16
#define DH_  64
#define DC_  256
#define NS_  512        // cache slots
#define KW_  64         // max write tokens
#define BS_  4096       // B_*S_
#define EPS_ 1e-5f

typedef __hip_bfloat16 bf16_t;
typedef __attribute__((ext_vector_type(16))) __bf16 v16bf;
typedef __attribute__((ext_vector_type(8)))  float  v8f;

struct __align__(16) U128 { unsigned int x, y, z, w; };
union FragU { U128 q[2]; v16bf v; };

// ---------------- WMMA fragment loads (CDNA5 16-bit layouts) ----------------
// A (16x32, MxK): lanes 0-15 row M=lane, K = {k0..k0+7, k0+16..k0+23};
//                 lanes 16-31 same rows, K shifted by +8.
__device__ inline v16bf load_a_frag(const bf16_t* A, int lda, int m, int k0, int lane) {
  const bf16_t* p = A + (size_t)m * lda + k0 + ((lane >> 4) << 3);
  FragU f;
  f.q[0] = *(const U128*)(p);
  f.q[1] = *(const U128*)(p + 16);
  return f.v;
}
// B (32x16, KxN) sourced from row-major [N,K]: lane = column N, K = {k0..k0+7,
// k0+16..k0+23}; lanes 16-31 K shifted by +8.
__device__ inline v16bf load_b_frag(const bf16_t* Bm, int ldb, int n, int k0, int lane) {
  const bf16_t* p = Bm + (size_t)n * ldb + k0 + ((lane >> 4) << 3);
  FragU f;
  f.q[0] = *(const U128*)(p);
  f.q[1] = *(const U128*)(p + 16);
  return f.v;
}

__device__ inline float gelu_f(float x) {
  return 0.5f * x * (1.0f + tanhf(0.7978845608f * (x + 0.044715f * x * x * x)));
}
__device__ inline float sigmoid_f(float x) { return 1.0f / (1.0f + __expf(-x)); }

// ---------------- generic bf16 WMMA GEMM: C = A * B^T (+epilogue) ----------------
// A: [M,K] bf16 row-major (lda); B: [N,K] bf16 row-major (ldb); C: [M,N] f32 (ldc).
// One wave computes a 32x64 tile: 2 A-fragments x 4 B-fragments -> 8 WMMAs/K-step.
// (A reused 4x, B reused 2x -> 8 WMMAs per 12 b128 loads.)
// Block = 8 waves stacked along M (256 rows). grid = (N/64, ceil(M/256), Z).
// act: 0=none 1=gelu 2=sigmoid; rowscale: per-row multiplier (applied before resid).
__global__ __launch_bounds__(256)
void gemm_bf16_nt(const bf16_t* __restrict__ A, const bf16_t* __restrict__ Bm,
                  const float* __restrict__ bias, const float* __restrict__ resid,
                  const float* __restrict__ rowscale,
                  float* __restrict__ C, bf16_t* __restrict__ Cbf,
                  int M, int N, int K, int lda, int ldb, int ldc,
                  long sA, long sB, long sC, int act)
{
  const int lane = threadIdx.x & 31;
  const int wid  = threadIdx.x >> 5;
  const int z    = blockIdx.z;
  A  += (size_t)z * sA;
  Bm += (size_t)z * sB;
  const size_t coff = (size_t)z * sC;

  const int m0 = (blockIdx.y * 8 + wid) * 32;
  const int n0 = blockIdx.x * 64;
  if (m0 >= M) return;            // uniform per wave: EXEC stays all-ones for WMMA

  int mA0 = m0 + (lane & 15);      if (mA0 > M - 1) mA0 = M - 1;
  int mA1 = m0 + 16 + (lane & 15); if (mA1 > M - 1) mA1 = M - 1;
  const int nb  = n0 + (lane & 15);
  const int nb0 = (nb      > N - 1) ? N - 1 : nb;
  const int nb1 = (nb + 16 > N - 1) ? N - 1 : nb + 16;
  const int nb2 = (nb + 32 > N - 1) ? N - 1 : nb + 32;
  const int nb3 = (nb + 48 > N - 1) ? N - 1 : nb + 48;

  v8f acc00 = {}, acc01 = {}, acc02 = {}, acc03 = {};
  v8f acc10 = {}, acc11 = {}, acc12 = {}, acc13 = {};
  for (int k0 = 0; k0 < K; k0 += 32) {
    v16bf a0 = load_a_frag(A, lda, mA0, k0, lane);
    v16bf a1 = load_a_frag(A, lda, mA1, k0, lane);
    v16bf b0 = load_b_frag(Bm, ldb, nb0, k0, lane);
    v16bf b1 = load_b_frag(Bm, ldb, nb1, k0, lane);
    v16bf b2 = load_b_frag(Bm, ldb, nb2, k0, lane);
    v16bf b3 = load_b_frag(Bm, ldb, nb3, k0, lane);
    acc00 = __builtin_amdgcn_wmma_f32_16x16x32_bf16(false, a0, false, b0, (short)0, acc00, false, false);
    acc01 = __builtin_amdgcn_wmma_f32_16x16x32_bf16(false, a0, false, b1, (short)0, acc01, false, false);
    acc02 = __builtin_amdgcn_wmma_f32_16x16x32_bf16(false, a0, false, b2, (short)0, acc02, false, false);
    acc03 = __builtin_amdgcn_wmma_f32_16x16x32_bf16(false, a0, false, b3, (short)0, acc03, false, false);
    acc10 = __builtin_amdgcn_wmma_f32_16x16x32_bf16(false, a1, false, b0, (short)0, acc10, false, false);
    acc11 = __builtin_amdgcn_wmma_f32_16x16x32_bf16(false, a1, false, b1, (short)0, acc11, false, false);
    acc12 = __builtin_amdgcn_wmma_f32_16x16x32_bf16(false, a1, false, b2, (short)0, acc12, false, false);
    acc13 = __builtin_amdgcn_wmma_f32_16x16x32_bf16(false, a1, false, b3, (short)0, acc13, false, false);
  }

  // C/D layout: lanes 0-15 -> N=lane, M=r ; lanes 16-31 -> N=lane-16, M=r+8
  const int mrow0 = m0 + ((lane >> 4) << 3);
  const int mrow1 = m0 + 16 + ((lane >> 4) << 3);
  const int col0  = n0 + (lane & 15);

#define EPILOGUE(ACC, MR, J)                                                 \
  {                                                                          \
    const int col = col0 + 16 * (J);                                         \
    if (col < N) {                                                           \
      const float bv = bias ? bias[col] : 0.0f;                              \
      _Pragma("unroll")                                                      \
      for (int r = 0; r < 8; ++r) {                                          \
        const int row = (MR) + r;                                            \
        if (row < M) {                                                       \
          float v = ACC[r] + bv;                                             \
          if (act == 1)      v = gelu_f(v);                                  \
          else if (act == 2) v = sigmoid_f(v);                               \
          if (rowscale) v *= rowscale[row];                                  \
          const size_t oidx = coff + (size_t)row * ldc + col;                \
          if (resid) v += resid[oidx];                                       \
          if (C)    C[oidx]   = v;                                           \
          if (Cbf)  Cbf[oidx] = __float2bfloat16(v);                         \
        }                                                                    \
      }                                                                      \
    }                                                                        \
  }
  EPILOGUE(acc00, mrow0, 0)
  EPILOGUE(acc01, mrow0, 1)
  EPILOGUE(acc02, mrow0, 2)
  EPILOGUE(acc03, mrow0, 3)
  EPILOGUE(acc10, mrow1, 0)
  EPILOGUE(acc11, mrow1, 1)
  EPILOGUE(acc12, mrow1, 2)
  EPILOGUE(acc13, mrow1, 3)
#undef EPILOGUE
}

// ---------------- LayerNorm (one 256-thread block per row) ----------------
__global__ __launch_bounds__(256)
void layernorm_kernel(const float* __restrict__ in, const float* __restrict__ g,
                      const float* __restrict__ b, float* __restrict__ out,
                      bf16_t* __restrict__ outbf, int Dm)
{
  __shared__ float red0[8], red1[8];
  __shared__ float s_mean, s_inv;
  const size_t row = blockIdx.x;
  const float* x = in + row * Dm;
  float s = 0.f, s2 = 0.f;
  for (int i = threadIdx.x; i < Dm; i += 256) { float v = x[i]; s += v; s2 += v * v; }
  for (int o = 16; o; o >>= 1) { s += __shfl_xor(s, o); s2 += __shfl_xor(s2, o); }
  if ((threadIdx.x & 31) == 0) { red0[threadIdx.x >> 5] = s; red1[threadIdx.x >> 5] = s2; }
  __syncthreads();
  if (threadIdx.x == 0) {
    float t0 = 0.f, t1 = 0.f;
    for (int k = 0; k < 8; ++k) { t0 += red0[k]; t1 += red1[k]; }
    float mean = t0 / Dm;
    s_mean = mean;
    s_inv  = rsqrtf(t1 / Dm - mean * mean + EPS_);
  }
  __syncthreads();
  const float mean = s_mean, inv = s_inv;
  for (int i = threadIdx.x; i < Dm; i += 256) {
    float v = (x[i] - mean) * inv * g[i] + b[i];
    if (out)   out[row * Dm + i]   = v;
    if (outbf) outbf[row * Dm + i] = __float2bfloat16(v);
  }
}

// ---------------- row softmax (scale pre, optional per-row post-multiplier) -----
__global__ __launch_bounds__(256)
void softmax_kernel(const float* __restrict__ in, float* __restrict__ out,
                    bf16_t* __restrict__ outbf, const float* __restrict__ rowmul,
                    float scale, int L)
{
  __shared__ float red[8];
  __shared__ float s_bc;
  const size_t row = blockIdx.x;
  const float* x = in + row * L;
  float mx = -3.4e38f;
  for (int i = threadIdx.x; i < L; i += 256) mx = fmaxf(mx, x[i] * scale);
  for (int o = 16; o; o >>= 1) mx = fmaxf(mx, __shfl_xor(mx, o));
  if ((threadIdx.x & 31) == 0) red[threadIdx.x >> 5] = mx;
  __syncthreads();
  if (threadIdx.x == 0) { float m = red[0]; for (int k = 1; k < 8; ++k) m = fmaxf(m, red[k]); s_bc = m; }
  __syncthreads();
  mx = s_bc;
  __syncthreads();
  float sum = 0.f;
  for (int i = threadIdx.x; i < L; i += 256) sum += __expf(x[i] * scale - mx);
  for (int o = 16; o; o >>= 1) sum += __shfl_xor(sum, o);
  if ((threadIdx.x & 31) == 0) red[threadIdx.x >> 5] = sum;
  __syncthreads();
  if (threadIdx.x == 0) { float t = 0.f; for (int k = 0; k < 8; ++k) t += red[k]; s_bc = 1.0f / t; }
  __syncthreads();
  const float inv = s_bc * (rowmul ? rowmul[row] : 1.0f);
  for (int i = threadIdx.x; i < L; i += 256) {
    float v = __expf(x[i] * scale - mx) * inv;
    if (out)   out[row * L + i]   = v;
    if (outbf) outbf[row * L + i] = __float2bfloat16(v);
  }
}

// ---------------- row dot with a single weight vector (+bias, opt sigmoid) ------
__global__ __launch_bounds__(256)
void rowdot_kernel(const float* __restrict__ in, const float* __restrict__ w,
                   const float* __restrict__ bptr, float* __restrict__ out,
                   int Dm, int sigm)
{
  __shared__ float red[8];
  const size_t row = blockIdx.x;
  const float* x = in + row * Dm;
  float s = 0.f;
  for (int i = threadIdx.x; i < Dm; i += 256) s += x[i] * w[i];
  for (int o = 16; o; o >>= 1) s += __shfl_xor(s, o);
  if ((threadIdx.x & 31) == 0) red[threadIdx.x >> 5] = s;
  __syncthreads();
  if (threadIdx.x == 0) {
    float t = 0.f;
    for (int k = 0; k < 8; ++k) t += red[k];
    t += bptr[0];
    out[row] = sigm ? sigmoid_f(t) : t;
  }
}

// ---------------- top-64 per batch (iterative argmax, lowest-index tie-break) ---
__global__ __launch_bounds__(256)
void topk64_kernel(const float* __restrict__ imp, int* __restrict__ idx, int Sn)
{
  __shared__ float sv[S_];
  __shared__ float rv[256];
  __shared__ int   ri[256];
  const int b = blockIdx.x;
  for (int i = threadIdx.x; i < Sn; i += 256) sv[i] = imp[(size_t)b * Sn + i];
  __syncthreads();
  for (int k = 0; k < KW_; ++k) {
    float best = -3.4e38f; int bi = Sn;
    for (int i = threadIdx.x; i < Sn; i += 256) {
      float v = sv[i];
      if (v > best || (v == best && i < bi)) { best = v; bi = i; }
    }
    rv[threadIdx.x] = best; ri[threadIdx.x] = bi;
    __syncthreads();
    if (threadIdx.x == 0) {
      float bb = -3.4e38f; int bbi = Sn;
      for (int t = 0; t < 256; ++t) {
        if (rv[t] > bb || (rv[t] == bb && ri[t] < bbi)) { bb = rv[t]; bbi = ri[t]; }
      }
      idx[b * KW_ + k] = bbi;
      sv[bbi] = -3.4e38f;
    }
    __syncthreads();
  }
}

// ---------------- small elementwise helpers ----------------
__global__ void f32_to_bf16_kernel(const float* __restrict__ in, bf16_t* __restrict__ out, size_t n) {
  size_t i = (size_t)blockIdx.x * 256 + threadIdx.x;
  if (i < n) out[i] = __float2bfloat16(in[i]);
}

// batched transpose f32[R,C] -> bf16[C,R]
__global__ void transpose_f2b_kernel(const float* __restrict__ in, bf16_t* __restrict__ out,
                                     int R, int Cc, long sIn, long sOut, size_t total) {
  size_t i = (size_t)blockIdx.x * 256 + threadIdx.x;
  if (i >= total) return;
  size_t per = (size_t)R * Cc;
  size_t z = i / per, rem = i % per;
  int r = (int)(rem / Cc), c = (int)(rem % Cc);
  out[(size_t)z * sOut + (size_t)c * R + r] = __float2bfloat16(in[(size_t)z * sIn + rem]);
}

__global__ void concat2_bf16_kernel(const float* __restrict__ a, const float* __restrict__ b,
                                    bf16_t* __restrict__ out, size_t n) {
  size_t i = (size_t)blockIdx.x * 256 + threadIdx.x;
  if (i >= n) return;
  size_t row = i / (2 * D_); int c = (int)(i % (2 * D_));
  float v = (c < D_) ? a[row * D_ + c] : b[row * D_ + (c - D_)];
  out[i] = __float2bfloat16(v);
}

__global__ void addgate_kernel(const float* __restrict__ h, const float* __restrict__ g,
                               const float* __restrict__ hp, float* __restrict__ out, size_t n) {
  size_t i = (size_t)blockIdx.x * 256 + threadIdx.x;
  if (i < n) out[i] = h[i] + g[i] * hp[i];
}

// qkv [BS,3D] f32 -> q,k bf16 [B,H,S,DH], v bf16 transposed [B,H,DH,S]
__global__ void split_qkv_kernel(const float* __restrict__ qkv, bf16_t* __restrict__ qh,
                                 bf16_t* __restrict__ kh, bf16_t* __restrict__ vT) {
  size_t i = (size_t)blockIdx.x * 256 + threadIdx.x;
  if (i >= (size_t)BS_ * D_) return;
  int d = (int)(i % D_); size_t bs = i / D_;
  int s = (int)(bs % S_); int b = (int)(bs / S_);
  int hh = d / DH_, dh = d % DH_;
  const float* base = qkv + bs * (3 * D_);
  size_t hb = (size_t)b * H_ + hh;
  qh[(hb * S_ + s) * DH_ + dh] = __float2bfloat16(base[d]);
  kh[(hb * S_ + s) * DH_ + dh] = __float2bfloat16(base[D_ + d]);
  vT[(hb * DH_ + dh) * S_ + s] = __float2bfloat16(base[2 * D_ + d]);
}

__global__ void gather_kernel(const float* __restrict__ h, const int* __restrict__ idx,
                              float* __restrict__ sel, bf16_t* __restrict__ selb) {
  size_t i = (size_t)blockIdx.x * 256 + threadIdx.x;
  if (i >= (size_t)B_ * KW_ * D_) return;
  int d = (int)(i % D_); size_t bk = i / D_;
  int k = (int)(bk % KW_); int b = (int)(bk / KW_);
  int s = idx[b * KW_ + k];
  float v = h[((size_t)b * S_ + s) * D_ + d];
  sel[i]  = v;
  selb[i] = __float2bfloat16(v);
}

// tot[b,n] = clip(sum_k w[b,k,n], 0, 1)
__global__ void totclip_kernel(const float* __restrict__ w, float* __restrict__ tot) {
  int i = blockIdx.x * 256 + threadIdx.x;
  if (i >= B_ * NS_) return;
  int b = i / NS_, n = i % NS_;
  float s = 0.f;
  for (int k = 0; k < KW_; ++k) s += w[((size_t)b * KW_ + k) * NS_ + n];
  tot[i] = fminf(fmaxf(s, 0.f), 1.f);
}

__global__ void cache_update_kernel(const float* __restrict__ cin, const float* __restrict__ delta,
                                    const float* __restrict__ tot, float* __restrict__ cout) {
  size_t i = (size_t)blockIdx.x * 256 + threadIdx.x;
  if (i >= (size_t)B_ * NS_ * DC_) return;
  float t = tot[i / DC_];
  cout[i] = cin[i] * (1.f - t) + delta[i];
}

// ---------------- host side ----------------
static inline void launch_gemm(hipStream_t st, const bf16_t* A, const bf16_t* Bm,
                               const float* bias, const float* resid, const float* rowscale,
                               float* C, bf16_t* Cbf, int M, int N, int K,
                               int lda, int ldb, int ldc,
                               long sA, long sB, long sC, int Z, int act)
{
  dim3 grid((N + 63) / 64, (M + 255) / 256, Z);
  gemm_bf16_nt<<<grid, dim3(256), 0, st>>>(A, Bm, bias, resid, rowscale, C, Cbf,
                                           M, N, K, lda, ldb, ldc, sA, sB, sC, act);
}

extern "C" void kernel_launch(void* const* d_in, const int* in_sizes, int n_in,
                              void* d_out, int out_size, void* d_ws, size_t ws_size,
                              hipStream_t stream)
{
  (void)in_sizes; (void)n_in; (void)out_size; (void)ws_size;
  const float* x      = (const float*)d_in[0];
  const float* cache0 = (const float*)d_in[1];
  const float* qkv_w  = (const float*)d_in[2];
  const float* qkv_b  = (const float*)d_in[3];
  const float* out_w  = (const float*)d_in[4];
  const float* out_b  = (const float*)d_in[5];
  const float* ffn1_w = (const float*)d_in[6];
  const float* ffn1_b = (const float*)d_in[7];
  const float* ffn2_w = (const float*)d_in[8];
  const float* ffn2_b = (const float*)d_in[9];
  const float* ln1_g  = (const float*)d_in[10];
  const float* ln1_b  = (const float*)d_in[11];
  const float* ln2_g  = (const float*)d_in[12];
  const float* ln2_b  = (const float*)d_in[13];
  const float* fb_w   = (const float*)d_in[14];
  const float* fb_b   = (const float*)d_in[15];
  const float* rq_w   = (const float*)d_in[16];
  const float* rq_b   = (const float*)d_in[17];
  const float* ro_w   = (const float*)d_in[18];
  const float* ro_b   = (const float*)d_in[19];
  const float* rg_w   = (const float*)d_in[20];
  const float* rg_b   = (const float*)d_in[21];
  const float* imp_w  = (const float*)d_in[22];
  const float* imp_b  = (const float*)d_in[23];
  const float* wk_w   = (const float*)d_in[24];
  const float* wk_b   = (const float*)d_in[25];
  const float* wv_w   = (const float*)d_in[26];
  const float* wv_b   = (const float*)d_in[27];
  const float* wg_w   = (const float*)d_in[28];
  const float* wg_b   = (const float*)d_in[29];

  char* base = (char*)d_ws;
  size_t off = 0;
  auto alloc = [&](size_t bytes) -> void* {
    void* p = base + off;
    off += (bytes + 255) & ~(size_t)255;
    return p;
  };

  // bf16 weights
  bf16_t* qkvw_b = (bf16_t*)alloc((size_t)3 * D_ * D_ * 2);
  bf16_t* outw_b = (bf16_t*)alloc((size_t)D_ * D_ * 2);
  bf16_t* f1w_b  = (bf16_t*)alloc((size_t)4 * D_ * D_ * 2);
  bf16_t* f2w_b  = (bf16_t*)alloc((size_t)4 * D_ * D_ * 2);
  bf16_t* fbw_b  = (bf16_t*)alloc((size_t)2 * D_ * D_ * 2);
  bf16_t* rqw_b  = (bf16_t*)alloc((size_t)DC_ * D_ * 2);
  bf16_t* row_b  = (bf16_t*)alloc((size_t)D_ * DC_ * 2);
  bf16_t* wkw_b  = (bf16_t*)alloc((size_t)DC_ * D_ * 2);
  bf16_t* wvw_b  = (bf16_t*)alloc((size_t)DC_ * D_ * 2);

  // big reusable region (48MB): {cat_bf + gbuf} | {qkv f32} | {ffn1 bf16}
  char*   bigA    = (char*)alloc((size_t)BS_ * 3 * D_ * 4);
  bf16_t* cat_bf  = (bf16_t*)bigA;                                   // 16MB
  float*  gbuf    = (float*)(bigA + (size_t)BS_ * 2 * D_ * 2);        // 16MB
  float*  qkvbuf  = (float*)bigA;                                     // 48MB
  bf16_t* f1_bf   = (bf16_t*)bigA;                                    // 32MB

  // reusable region (24MB): {mem-read scores + att} | {per-head scores + att}
  char*   R1       = (char*)alloc((size_t)S_ * S_ * 4 + (size_t)S_ * S_ * 2);
  float*  scoresmr = (float*)R1;                                      // 8MB
  bf16_t* attmr_bf = (bf16_t*)(R1 + (size_t)16 * 1024 * 1024);        // 4MB
  float*  scbuf    = (float*)R1;                                      // 16MB
  bf16_t* attb_bf  = (bf16_t*)(R1 + (size_t)16 * 1024 * 1024);        // 8MB

  float*  hcmpA  = (float*)alloc((size_t)BS_ * D_ * 4);
  float*  hcmpB  = (float*)alloc((size_t)BS_ * D_ * 4);
  float*  hwork  = (float*)alloc((size_t)BS_ * D_ * 4);
  bf16_t* h_bf   = (bf16_t*)alloc((size_t)BS_ * D_ * 2);
  float*  h_enh  = (float*)alloc((size_t)BS_ * D_ * 4);
  float*  x1buf  = (float*)alloc((size_t)BS_ * D_ * 4);
  float*  x2buf  = (float*)alloc((size_t)BS_ * D_ * 4);
  float*  x3buf  = (float*)alloc((size_t)BS_ * D_ * 4);
  bf16_t* x2_bf  = (bf16_t*)alloc((size_t)BS_ * D_ * 2);
  bf16_t* xn_bf  = (bf16_t*)alloc((size_t)BS_ * D_ * 2);
  bf16_t* q_bf   = (bf16_t*)alloc((size_t)BS_ * DC_ * 2);
  bf16_t* r_bf   = (bf16_t*)alloc((size_t)BS_ * DC_ * 2);
  float*  gate_r = (float*)alloc((size_t)BS_ * 4);
  bf16_t* qh     = (bf16_t*)alloc((size_t)B_ * H_ * S_ * DH_ * 2);
  bf16_t* kh     = (bf16_t*)alloc((size_t)B_ * H_ * S_ * DH_ * 2);
  bf16_t* vT     = (bf16_t*)alloc((size_t)B_ * H_ * DH_ * S_ * 2);
  bf16_t* o_bf   = (bf16_t*)alloc((size_t)BS_ * D_ * 2);
  bf16_t* cache_bf  = (bf16_t*)alloc((size_t)B_ * NS_ * DC_ * 2);
  bf16_t* cacheT_bf = (bf16_t*)alloc((size_t)B_ * DC_ * NS_ * 2);
  float*  cacheN0   = (float*)alloc((size_t)B_ * NS_ * DC_ * 4);
  float*  cacheN1   = (float*)alloc((size_t)B_ * NS_ * DC_ * 4);
  float*  impbuf  = (float*)alloc((size_t)BS_ * 4);
  int*    idxbuf  = (int*)alloc((size_t)B_ * KW_ * 4);
  float*  selbuf  = (float*)alloc((size_t)B_ * KW_ * D_ * 4);
  bf16_t* sel_bf  = (bf16_t*)alloc((size_t)B_ * KW_ * D_ * 2);
  bf16_t* keys_bf = (bf16_t*)alloc((size_t)B_ * KW_ * DC_ * 2);
  float*  valsbuf = (float*)alloc((size_t)B_ * KW_ * DC_ * 4);
  float*  gatew   = (float*)alloc((size_t)B_ * KW_ * 4);
  float*  routebuf= (float*)alloc((size_t)B_ * KW_ * NS_ * 4);
  float*  wbuf    = (float*)alloc((size_t)B_ * KW_ * NS_ * 4);
  bf16_t* wT_bf   = (bf16_t*)alloc((size_t)B_ * NS_ * KW_ * 2);
  bf16_t* valsT_bf= (bf16_t*)alloc((size_t)B_ * DC_ * KW_ * 2);
  float*  deltabuf= (float*)alloc((size_t)B_ * NS_ * DC_ * 4);
  float*  totbuf  = (float*)alloc((size_t)B_ * NS_ * 4);

  auto conv = [&](const float* src, bf16_t* dst, size_t n) {
    f32_to_bf16_kernel<<<dim3((unsigned)((n + 255) / 256)), dim3(256), 0, stream>>>(src, dst, n);
  };

  // weights -> bf16 (once per call)
  conv(qkv_w,  qkvw_b, (size_t)3 * D_ * D_);
  conv(out_w,  outw_b, (size_t)D_ * D_);
  conv(ffn1_w, f1w_b,  (size_t)4 * D_ * D_);
  conv(ffn2_w, f2w_b,  (size_t)4 * D_ * D_);
  conv(fb_w,   fbw_b,  (size_t)2 * D_ * D_);
  conv(rq_w,   rqw_b,  (size_t)DC_ * D_);
  conv(ro_w,   row_b,  (size_t)D_ * DC_);
  conv(wk_w,   wkw_b,  (size_t)DC_ * D_);
  conv(wv_w,   wvw_b,  (size_t)DC_ * D_);

  const float* h      = x;
  const float* hprev  = nullptr;
  const float* cin    = cache0;
  float*       cout_p = cacheN0;
  float*       hcmp_t[2] = { hcmpA, hcmpB };

  for (int it = 0; it < 2; ++it) {
    // ---- iteration feedback gate ----
    if (it > 0) {
      size_t ncat = (size_t)BS_ * 2 * D_;
      concat2_bf16_kernel<<<dim3((unsigned)((ncat + 255) / 256)), 256, 0, stream>>>(h, hprev, cat_bf, ncat);
      launch_gemm(stream, cat_bf, fbw_b, fb_b, nullptr, nullptr, gbuf, nullptr,
                  BS_, D_, 2 * D_, 2 * D_, 2 * D_, D_, 0, 0, 0, 1, /*sigmoid*/2);
      size_t nel = (size_t)BS_ * D_;
      addgate_kernel<<<dim3((unsigned)((nel + 255) / 256)), 256, 0, stream>>>(h, gbuf, hprev, hwork, nel);
      h = hwork;
    }

    // ---- cache views for this iteration ----
    conv(cin, cache_bf, (size_t)B_ * NS_ * DC_);
    {
      size_t tot = (size_t)B_ * NS_ * DC_;
      transpose_f2b_kernel<<<dim3((unsigned)((tot + 255) / 256)), 256, 0, stream>>>(
          cin, cacheT_bf, NS_, DC_, (long)NS_ * DC_, (long)DC_ * NS_, tot);
    }
    conv(h, h_bf, (size_t)BS_ * D_);

    // ---- memory read ----
    launch_gemm(stream, h_bf, rqw_b, rq_b, nullptr, nullptr, nullptr, q_bf,
                BS_, DC_, D_, D_, D_, DC_, 0, 0, 0, 1, 0);
    launch_gemm(stream, q_bf, cache_bf, nullptr, nullptr, nullptr, scoresmr, nullptr,
                S_, NS_, DC_, DC_, DC_, NS_,
                (long)S_ * DC_, (long)NS_ * DC_, (long)S_ * NS_, B_, 0);
    softmax_kernel<<<dim3(B_ * S_), 256, 0, stream>>>(scoresmr, nullptr, attmr_bf, nullptr, 1.0f / 16.0f, NS_);
    launch_gemm(stream, attmr_bf, cacheT_bf, nullptr, nullptr, nullptr, nullptr, r_bf,
                S_, DC_, NS_, NS_, NS_, DC_,
                (long)S_ * NS_, (long)DC_ * NS_, (long)S_ * DC_, B_, 0);
    rowdot_kernel<<<dim3(BS_), 256, 0, stream>>>(h, rg_w, rg_b, gate_r, D_, 1);
    launch_gemm(stream, r_bf, row_b, ro_b, h /*resid*/, gate_r /*rowscale*/, h_enh, nullptr,
                BS_, D_, DC_, DC_, DC_, D_, 0, 0, 0, 1, 0);

    // ---- compute block ----
    layernorm_kernel<<<dim3(BS_), 256, 0, stream>>>(h_enh, ln1_g, ln1_b, nullptr, xn_bf, D_);
    launch_gemm(stream, xn_bf, qkvw_b, qkv_b, nullptr, nullptr, qkvbuf, nullptr,
                BS_, 3 * D_, D_, D_, D_, 3 * D_, 0, 0, 0, 1, 0);
    {
      size_t tot = (size_t)BS_ * D_;
      split_qkv_kernel<<<dim3((unsigned)((tot + 255) / 256)), 256, 0, stream>>>(qkvbuf, qh, kh, vT);
    }
    for (int bh = 0; bh < B_ * H_; ++bh) {
      const bf16_t* qs = qh + (size_t)bh * S_ * DH_;
      const bf16_t* ks = kh + (size_t)bh * S_ * DH_;
      const bf16_t* vs = vT + (size_t)bh * DH_ * S_;
      launch_gemm(stream, qs, ks, nullptr, nullptr, nullptr, scbuf, nullptr,
                  S_, S_, DH_, DH_, DH_, S_, 0, 0, 0, 1, 0);
      softmax_kernel<<<dim3(S_), 256, 0, stream>>>(scbuf, nullptr, attb_bf, nullptr, 0.125f, S_);
      int b2 = bh / H_, hh = bh % H_;
      launch_gemm(stream, attb_bf, vs, nullptr, nullptr, nullptr, nullptr,
                  o_bf + (size_t)b2 * S_ * D_ + (size_t)hh * DH_,
                  S_, DH_, S_, S_, S_, D_, 0, 0, 0, 1, 0);
    }
    launch_gemm(stream, o_bf, outw_b, out_b, h_enh /*resid*/, nullptr, x1buf, nullptr,
                BS_, D_, D_, D_, D_, D_, 0, 0, 0, 1, 0);
    layernorm_kernel<<<dim3(BS_), 256, 0, stream>>>(x1buf, ln1_g, ln1_b, x2buf, x2_bf, D_);
    launch_gemm(stream, x2_bf, f1w_b, ffn1_b, nullptr, nullptr, nullptr, f1_bf,
                BS_, 4 * D_, D_, D_, D_, 4 * D_, 0, 0, 0, 1, /*gelu*/1);
    launch_gemm(stream, f1_bf, f2w_b, ffn2_b, x2buf /*resid*/, nullptr, x3buf, nullptr,
                BS_, D_, 4 * D_, 4 * D_, 4 * D_, D_, 0, 0, 0, 1, 0);
    float* hcmp = hcmp_t[it];
    layernorm_kernel<<<dim3(BS_), 256, 0, stream>>>(x3buf, ln2_g, ln2_b, hcmp, nullptr, D_);

    // ---- memory write ----
    rowdot_kernel<<<dim3(BS_), 256, 0, stream>>>(hcmp, imp_w, imp_b, impbuf, D_, 0);
    topk64_kernel<<<dim3(B_), 256, 0, stream>>>(impbuf, idxbuf, S_);
    {
      size_t tot = (size_t)B_ * KW_ * D_;
      gather_kernel<<<dim3((unsigned)((tot + 255) / 256)), 256, 0, stream>>>(hcmp, idxbuf, selbuf, sel_bf);
    }
    launch_gemm(stream, sel_bf, wkw_b, wk_b, nullptr, nullptr, nullptr, keys_bf,
                B_ * KW_, DC_, D_, D_, D_, DC_, 0, 0, 0, 1, 0);
    launch_gemm(stream, sel_bf, wvw_b, wv_b, nullptr, nullptr, valsbuf, nullptr,
                B_ * KW_, DC_, D_, D_, D_, DC_, 0, 0, 0, 1, 0);
    rowdot_kernel<<<dim3(B_ * KW_), 256, 0, stream>>>(selbuf, wg_w, wg_b, gatew, D_, 1);
    launch_gemm(stream, keys_bf, cache_bf, nullptr, nullptr, nullptr, routebuf, nullptr,
                KW_, NS_, DC_, DC_, DC_, NS_,
                (long)KW_ * DC_, (long)NS_ * DC_, (long)KW_ * NS_, B_, 0);
    softmax_kernel<<<dim3(B_ * KW_), 256, 0, stream>>>(routebuf, wbuf, nullptr, gatew, 1.0f / 16.0f, NS_);
    {
      size_t tw = (size_t)B_ * KW_ * NS_;
      transpose_f2b_kernel<<<dim3((unsigned)((tw + 255) / 256)), 256, 0, stream>>>(
          wbuf, wT_bf, KW_, NS_, (long)KW_ * NS_, (long)NS_ * KW_, tw);
      size_t tv = (size_t)B_ * KW_ * DC_;
      transpose_f2b_kernel<<<dim3((unsigned)((tv + 255) / 256)), 256, 0, stream>>>(
          valsbuf, valsT_bf, KW_, DC_, (long)KW_ * DC_, (long)DC_ * KW_, tv);
    }
    launch_gemm(stream, wT_bf, valsT_bf, nullptr, nullptr, nullptr, deltabuf, nullptr,
                NS_, DC_, KW_, KW_, KW_, DC_,
                (long)NS_ * KW_, (long)DC_ * KW_, (long)NS_ * DC_, B_, 0);
    totclip_kernel<<<dim3((B_ * NS_ + 255) / 256), 256, 0, stream>>>(wbuf, totbuf);
    {
      size_t tot = (size_t)B_ * NS_ * DC_;
      cache_update_kernel<<<dim3((unsigned)((tot + 255) / 256)), 256, 0, stream>>>(cin, deltabuf, totbuf, cout_p);
    }

    hprev = hcmp;
    h = hcmp;
    cin = cout_p;
    cout_p = cacheN1;
  }

  // outputs: h [B,S,D] then cache [B,N,DC]
  hipMemcpyAsync(d_out, h, (size_t)BS_ * D_ * sizeof(float), hipMemcpyDeviceToDevice, stream);
  hipMemcpyAsync((float*)d_out + (size_t)BS_ * D_, cin,
                 (size_t)B_ * NS_ * DC_ * sizeof(float), hipMemcpyDeviceToDevice, stream);
}
